// LlamaAttention_70325794504845
// MI455X (gfx1250) — matrix operand order repaired and checked by reading, not compiled, not run on hardware
//
#include <hip/hip_runtime.h>
#include <hip/hip_bf16.h>

// ---------------- problem constants ----------------
constexpr int B_  = 2;
constexpr int S_  = 2048;
constexpr int D_  = 2048;
constexpr int H_  = 16;
constexpr int DH_ = 128;
constexpr int BS_ = B_ * S_;          // 4096 rows

typedef __attribute__((ext_vector_type(16))) __bf16 v16bf;
typedef __attribute__((ext_vector_type(8)))  __bf16 v8bf;
typedef __attribute__((ext_vector_type(8)))  float  v8f;

// ---------------- WMMA helpers ----------------
__device__ __forceinline__ v8f wmma_bf16(v16bf a, v16bf b, v8f c) {
    // D = A(16x32 bf16) * B(32x16 bf16) + C(16x16 f32)
    return __builtin_amdgcn_wmma_f32_16x16x32_bf16(
        /*neg_a=*/false, a, /*neg_b=*/false, b,
        /*c_mod=*/(short)0, c, /*reuse_a=*/false, /*reuse_b=*/false);
}

// A fragment (16x32, row-major rows): lane half hf, element e ->
//   K = (e>=8)*16 + hf*8 + (e&7)   => two contiguous 8-elem (16B) runs.
__device__ __forceinline__ v16bf load_a(const __bf16* row, int hf) {
    union { v16bf v; v8bf h[2]; } u;
    u.h[0] = *(const v8bf*)(row + hf * 8);
    u.h[1] = *(const v8bf*)(row + 16 + hf * 8);
    return u.v;
}

// B fragment (32x16, column `col` contiguous over K): lane half hf,
//   element e -> K = (e>=8)*16 + hf*8 + (e&7)   => two contiguous 8-elem (16B) runs.
__device__ __forceinline__ v16bf load_b(const __bf16* col, int hf) {
    union { v16bf v; v8bf h[2]; } u;
    u.h[0] = *(const v8bf*)(col + hf * 8);
    u.h[1] = *(const v8bf*)(col + 16 + hf * 8);
    return u.v;
}

// ---------------- CDNA5 async copy: global -> LDS, 16B per lane ----------
// VDST = VGPR with LDS byte offset (low 32 bits of generic shared pointer),
// VADDR = 64-bit global address. Tracked by ASYNCcnt.
__device__ __forceinline__ void async_copy_b128(const __bf16* g, void* lds) {
    const uint32_t lds_off = (uint32_t)(uintptr_t)lds;
    const uint64_t gaddr   = (uint64_t)(uintptr_t)g;
    asm volatile("global_load_async_to_lds_b128 %0, %1, off"
                 :: "v"(lds_off), "v"(gaddr) : "memory");
}

// ---------------- kernel 0: f32 -> bf16 convert ----------------
__global__ void cvt_kernel(const float* __restrict__ src,
                           __bf16* __restrict__ dst, int n) {
    int i = blockIdx.x * 256 + threadIdx.x;
    if (i < n) dst[i] = (__bf16)src[i];
}

// ---------------- shared GEMM body (64 x 256 block tile) ----------------
// block = 256 threads = 8 waves as 2x4; wave tile 32x64 (8 accumulators).
// Per 32-wide k-step: stage A(64x32) + B(256x32) bf16 into double-buffered
// LDS with async copies (5 x 16B per thread), overlap with 8 WMMAs/wave.
struct GemmCtx {
    v8f acc[8];           // [s*4+j] : s = M-subtile (2), j = N-subtile (4)
};

// branch-free staging: 1 A-chunk + 4 B-chunks of 16B per thread
__device__ __forceinline__ void stage_tiles(const __bf16* __restrict__ Abase,
                                            const __bf16* __restrict__ Bbase,
                                            int k0, int tid,
                                            __bf16 (*sA)[32],
                                            __bf16 (*sB)[32]) {
    {
        const int row = tid >> 2, ko = (tid & 3) * 8;        // A: 64x32
        async_copy_b128(Abase + (size_t)row * D_ + k0 + ko, &sA[row][ko]);
    }
#pragma unroll
    for (int i = 0; i < 4; ++i) {                            // B: 256x32
        const int c = tid + 256 * i;
        const int row = c >> 2, ko = (c & 3) * 8;
        async_copy_b128(Bbase + (size_t)row * D_ + k0 + ko, &sB[row][ko]);
    }
}

// one k-step of compute: issue all ds loads up front, then 8-WMMA burst
__device__ __forceinline__ void gemm_step(__bf16 (*sA)[32], __bf16 (*sB)[32],
                                          int wr, int wc, int hf, int ln,
                                          GemmCtx& ctx) {
    v16bf af[2], bfr[4];
#pragma unroll
    for (int s = 0; s < 2; ++s) {
        const int row = wr * 32 + s * 16 + ln;
        union { v16bf v; v8bf h[2]; } u;
        u.h[0] = *(const v8bf*)&sA[row][hf * 8];
        u.h[1] = *(const v8bf*)&sA[row][16 + hf * 8];
        af[s] = u.v;
    }
#pragma unroll
    for (int j = 0; j < 4; ++j) {
        const int col = wc * 64 + j * 16 + ln;
        union { v16bf v; v8bf h[2]; } u;
        u.h[0] = *(const v8bf*)&sB[col][hf * 8];
        u.h[1] = *(const v8bf*)&sB[col][16 + hf * 8];
        bfr[j] = u.v;
    }
#pragma unroll
    for (int j = 0; j < 4; ++j)
#pragma unroll
        for (int s = 0; s < 2; ++s)
            ctx.acc[s * 4 + j] = wmma_bf16(af[s], bfr[j], ctx.acc[s * 4 + j]);
}

__device__ __forceinline__ void gemm_body(const __bf16* __restrict__ Abase,
                                          const __bf16* __restrict__ Bbase,
                                          __bf16 (*shA)[64][32],
                                          __bf16 (*shB)[256][32],
                                          int tid, GemmCtx& ctx) {
    const int wave = tid >> 5, lane = tid & 31;
    const int hf = lane >> 4, ln = lane & 15;
    const int wr = wave >> 2, wc = wave & 3;      // 2 x 4 waves

    v8f z8 = {};
#pragma unroll
    for (int i = 0; i < 8; ++i) ctx.acc[i] = z8;

    stage_tiles(Abase, Bbase, 0, tid, shA[0], shB[0]);
    int cur = 0;
    // pipelined main loop (last iteration peeled -> straight-line body)
    for (int k0 = 0; k0 < D_ - 32; k0 += 32) {
        stage_tiles(Abase, Bbase, k0 + 32, tid, shA[cur ^ 1], shB[cur ^ 1]);
        asm volatile("s_wait_asynccnt 0x5" ::: "memory");  // prev stage done
        __syncthreads();
        gemm_step(shA[cur], shB[cur], wr, wc, hf, ln, ctx);
        __syncthreads();                                   // readers done
        cur ^= 1;
    }
    asm volatile("s_wait_asynccnt 0x0" ::: "memory");
    __syncthreads();
    gemm_step(shA[cur], shB[cur], wr, wc, hf, ln, ctx);
}

// ---------------- kernel 1: fused QKV projection GEMM ----------------
// grid = (BS/64, D/256, 3); block = 256
__global__ __launch_bounds__(256)
void proj_qkv_kernel(const __bf16* __restrict__ xb,
                     const __bf16* __restrict__ wqkv,   // Wq,Wk,Wv contiguous
                     __bf16* __restrict__ Qb,           // [B,H,S,DH]
                     __bf16* __restrict__ Kb,           // [B,H,S,DH]
                     __bf16* __restrict__ Vtb) {        // [B,H,DH,S]
    __shared__ alignas(16) __bf16 shA[2][64][32];       //  8 KB
    __shared__ alignas(16) __bf16 shB[2][256][32];      // 32 KB

    const int tid  = threadIdx.x;
    const int wave = tid >> 5, lane = tid & 31;
    const int hf = lane >> 4, ln = lane & 15;
    const int wr = wave >> 2, wc = wave & 3;
    const int Mblk = blockIdx.x * 64;
    const int Nblk = blockIdx.y * 256;
    const int z = blockIdx.z;

    const __bf16* W = wqkv + (size_t)z * D_ * D_;
    GemmCtx ctx;
    gemm_body(xb + (size_t)Mblk * D_, W + (size_t)Nblk * D_, shA, shB, tid, ctx);

    // epilogue: scatter into head-major layouts (V transposed)
#pragma unroll
    for (int s = 0; s < 2; ++s) {
#pragma unroll
        for (int r = 0; r < 8; ++r) {
            const int row_g = Mblk + wr * 32 + s * 16 + hf * 8 + r;
            const int bidx  = row_g >> 11;            // / S
            const int sq    = row_g & (S_ - 1);
#pragma unroll
            for (int j = 0; j < 4; ++j) {
                const int n  = Nblk + wc * 64 + j * 16 + ln;
                const int h  = n >> 7, dh = n & (DH_ - 1);
                const __bf16 bv = (__bf16)ctx.acc[s * 4 + j][r];
                if (z == 0)
                    Qb[(((size_t)bidx * H_ + h) * S_ + sq) * DH_ + dh] = bv;
                else if (z == 1)
                    Kb[(((size_t)bidx * H_ + h) * S_ + sq) * DH_ + dh] = bv;
                else
                    Vtb[(((size_t)bidx * H_ + h) * DH_ + dh) * S_ + sq] = bv;
            }
        }
    }
}

// ---------------- kernel 2: RoPE on Q and K (in place, race-free) ---------
__global__ void rope_kernel(__bf16* __restrict__ Qb, __bf16* __restrict__ Kb,
                            const float* __restrict__ cosb,
                            const float* __restrict__ sinb) {
    const size_t i   = (size_t)blockIdx.x * 256 + threadIdx.x;
    const int dh     = (int)(i & 63);
    const size_t rem = i >> 6;
    const int s      = (int)(rem & (S_ - 1));
    const size_t bh  = rem >> 11;
    const size_t base = (bh * S_ + s) * DH_;

    const float c1 = cosb[(size_t)s * DH_ + dh];
    const float s1 = sinb[(size_t)s * DH_ + dh];
    const float c2 = cosb[(size_t)s * DH_ + dh + 64];
    const float s2 = sinb[(size_t)s * DH_ + dh + 64];

    float q1 = (float)Qb[base + dh], q2 = (float)Qb[base + dh + 64];
    float k1 = (float)Kb[base + dh], k2 = (float)Kb[base + dh + 64];
    Qb[base + dh]      = (__bf16)(q1 * c1 - q2 * s1);
    Qb[base + dh + 64] = (__bf16)(q2 * c2 + q1 * s2);
    Kb[base + dh]      = (__bf16)(k1 * c1 - k2 * s1);
    Kb[base + dh + 64] = (__bf16)(k2 * c2 + k1 * s2);
}

// ---------------- kernel 3: causal flash attention ----------------
// grid = (S/128, B*H); block = 256 (8 waves, 16 query rows each)
__global__ __launch_bounds__(256)
void fa_kernel(const __bf16* __restrict__ Qb,   // [B,H,S,DH]
               const __bf16* __restrict__ Kb,   // [B,H,S,DH]
               const __bf16* __restrict__ Vtb,  // [B,H,DH,S]
               __bf16* __restrict__ valsb) {    // [B,S,D]
    __shared__ __bf16 lds_p[8][16][32];         // per-wave 16x32 P tile

    const int tid  = threadIdx.x;
    const int wave = tid >> 5, lane = tid & 31;
    const int hf   = lane >> 4, ln = lane & 15;
    const int bh   = blockIdx.y;
    const int b    = bh >> 4, h = bh & (H_ - 1);

    const __bf16* Q  = Qb  + (size_t)bh * S_ * DH_;
    const __bf16* Kp = Kb  + (size_t)bh * S_ * DH_;
    const __bf16* Vt = Vtb + (size_t)bh * DH_ * S_;

    const int qbase = blockIdx.x * 128 + wave * 16;

    v16bf qf[4];
    const __bf16* qrow = Q + (size_t)(qbase + ln) * DH_;
#pragma unroll
    for (int t = 0; t < 4; ++t) qf[t] = load_a(qrow + 32 * t, hf);

    v8f z8 = {};
    v8f acc[8];
#pragma unroll
    for (int d = 0; d < 8; ++d) acc[d] = z8;
    float mst[8], lst[8];
#pragma unroll
    for (int r = 0; r < 8; ++r) { mst[r] = -3.0e38f; lst[r] = 0.0f; }

    const float scale = 0.08838834764831845f;    // 1/sqrt(128)
    const int ktiles  = blockIdx.x * 4 + 4;      // causal upper bound

    for (int kt = 0; kt < ktiles; ++kt) {
        const int kbase = kt * 32;

        // prefetch next K tile rows into GL2 ahead of the fragment loads
        __builtin_prefetch(Kp + (size_t)(kbase + 32 + ln) * DH_, 0, 1);
        __builtin_prefetch(Vt + (size_t)(ln * 8) * S_ + kbase + 32, 0, 1);

        // ---- S = Q * K^T : contraction over DH (4 k-steps, 2 n-subtiles)
        v8f c0 = z8, c1 = z8;
#pragma unroll
        for (int t = 0; t < 4; ++t) {
            v16bf b0 = load_b(Kp + (size_t)(kbase + ln) * DH_ + 32 * t, hf);
            v16bf b1 = load_b(Kp + (size_t)(kbase + 16 + ln) * DH_ + 32 * t, hf);
            c0 = wmma_bf16(qf[t], b0, c0);
            c1 = wmma_bf16(qf[t], b1, c1);
        }

        // ---- scale, causal mask, online softmax update
        float p0s[8], p1s[8];
#pragma unroll
        for (int r = 0; r < 8; ++r) {
            const int mg = qbase + hf * 8 + r;
            float s0 = c0[r] * scale;
            float s1 = c1[r] * scale;
            if (kbase + ln > mg)      s0 = -1e30f;
            if (kbase + 16 + ln > mg) s1 = -1e30f;
            float lm = fmaxf(s0, s1);
#pragma unroll
            for (int m = 1; m < 16; m <<= 1)
                lm = fmaxf(lm, __shfl_xor(lm, m, 32));
            const float mnew = fmaxf(mst[r], lm);
            const float sc   = __expf(mst[r] - mnew);
            const float p0   = __expf(s0 - mnew);
            const float p1   = __expf(s1 - mnew);
            float rs = p0 + p1;
#pragma unroll
            for (int m = 1; m < 16; m <<= 1)
                rs += __shfl_xor(rs, m, 32);
            lst[r] = lst[r] * sc + rs;
            mst[r] = mnew;
#pragma unroll
            for (int d = 0; d < 8; ++d) acc[d][r] = acc[d][r] * sc;
            p0s[r] = p0; p1s[r] = p1;
        }

        // ---- relayout P: C-layout -> row-major LDS -> A-layout fragment
#pragma unroll
        for (int r = 0; r < 8; ++r) {
            lds_p[wave][hf * 8 + r][ln]      = (__bf16)p0s[r];
            lds_p[wave][hf * 8 + r][16 + ln] = (__bf16)p1s[r];
        }
        asm volatile("s_wait_dscnt 0" ::: "memory");   // wave-local LDS RAW
        union { v16bf v; v8bf hh[2]; } pu;
        pu.hh[0] = *(const v8bf*)&lds_p[wave][ln][hf * 8];
        pu.hh[1] = *(const v8bf*)&lds_p[wave][ln][16 + hf * 8];
        const v16bf pf = pu.v;
        asm volatile("s_wait_dscnt 0" ::: "memory");   // before next-iter stores

        // ---- O += P * V : contraction over 32 keys, 8 dh-subtiles
#pragma unroll
        for (int d = 0; d < 8; ++d) {
            v16bf vf = load_b(Vt + (size_t)(d * 16 + ln) * S_ + kbase, hf);
            acc[d] = wmma_bf16(pf, vf, acc[d]);
        }
    }

    // ---- epilogue: normalize and scatter into [B,S,D]
#pragma unroll
    for (int r = 0; r < 8; ++r) {
        const int sq    = qbase + hf * 8 + r;
        const float inv = 1.0f / lst[r];
#pragma unroll
        for (int d = 0; d < 8; ++d) {
            const int dh = d * 16 + ln;
            valsb[((size_t)b * S_ + sq) * D_ + h * DH_ + dh] =
                (__bf16)(acc[d][r] * inv);
        }
    }
}

// ---------------- kernel 4: output projection (f32 out) ----------------
// grid = (BS/64, D/256); block = 256
__global__ __launch_bounds__(256)
void proj_o_kernel(const __bf16* __restrict__ vb,
                   const __bf16* __restrict__ wob,
                   float* __restrict__ out) {
    __shared__ alignas(16) __bf16 shA[2][64][32];
    __shared__ alignas(16) __bf16 shB[2][256][32];

    const int tid  = threadIdx.x;
    const int wave = tid >> 5, lane = tid & 31;
    const int hf = lane >> 4, ln = lane & 15;
    const int wr = wave >> 2, wc = wave & 3;
    const int Mblk = blockIdx.x * 64;
    const int Nblk = blockIdx.y * 256;

    GemmCtx ctx;
    gemm_body(vb + (size_t)Mblk * D_, wob + (size_t)Nblk * D_, shA, shB, tid, ctx);

#pragma unroll
    for (int s = 0; s < 2; ++s) {
#pragma unroll
        for (int r = 0; r < 8; ++r) {
            const size_t row_g = Mblk + wr * 32 + s * 16 + hf * 8 + r;
#pragma unroll
            for (int j = 0; j < 4; ++j) {
                const int n = Nblk + wc * 64 + j * 16 + ln;
                out[row_g * D_ + n] = ctx.acc[s * 4 + j][r];
            }
        }
    }
}

// ---------------- host launcher ----------------
extern "C" void kernel_launch(void* const* d_in, const int* in_sizes, int n_in,
                              void* d_out, int out_size, void* d_ws, size_t ws_size,
                              hipStream_t stream) {
    const float* x    = (const float*)d_in[0];
    const float* cosb = (const float*)d_in[1];
    const float* sinb = (const float*)d_in[2];
    const float* Wq   = (const float*)d_in[3];
    const float* Wk   = (const float*)d_in[4];
    const float* Wv   = (const float*)d_in[5];
    const float* Wo   = (const float*)d_in[6];

    char* ws = (char*)d_ws;
    const size_t MB = 1024u * 1024u;
    __bf16* xb   = (__bf16*)(ws +  0 * MB);   // 16 MB  x bf16 [BS, D]
    __bf16* wqb  = (__bf16*)(ws + 16 * MB);   //  8 MB  Wq bf16 (Wq,Wk,Wv contiguous)
    __bf16* wkb  = (__bf16*)(ws + 24 * MB);
    __bf16* wvb  = (__bf16*)(ws + 32 * MB);
    __bf16* wob  = (__bf16*)(ws + 40 * MB);
    __bf16* Qb   = (__bf16*)(ws + 48 * MB);   // 16 MB  [B,H,S,DH]
    __bf16* Kb   = (__bf16*)(ws + 64 * MB);   // 16 MB  [B,H,S,DH]
    __bf16* Vtb  = (__bf16*)(ws + 80 * MB);   // 16 MB  [B,H,DH,S]
    __bf16* vals = (__bf16*)(ws + 96 * MB);   // 16 MB  [B,S,D]

    const int nx = BS_ * D_;                  // 8388608
    const int nw = D_ * D_;                   // 4194304

    cvt_kernel<<<nx / 256, 256, 0, stream>>>(x,  xb,  nx);
    cvt_kernel<<<nw / 256, 256, 0, stream>>>(Wq, wqb, nw);
    cvt_kernel<<<nw / 256, 256, 0, stream>>>(Wk, wkb, nw);
    cvt_kernel<<<nw / 256, 256, 0, stream>>>(Wv, wvb, nw);
    cvt_kernel<<<nw / 256, 256, 0, stream>>>(Wo, wob, nw);

    proj_qkv_kernel<<<dim3(BS_ / 64, D_ / 256, 3), 256, 0, stream>>>(
        xb, wqb, Qb, Kb, Vtb);

    rope_kernel<<<(B_ * H_ * S_ * 64) / 256, 256, 0, stream>>>(
        Qb, Kb, cosb, sinb);

    fa_kernel<<<dim3(S_ / 128, B_ * H_), 256, 0, stream>>>(Qb, Kb, Vtb, vals);

    proj_o_kernel<<<dim3(BS_ / 64, D_ / 256), 256, 0, stream>>>(
        vals, wob, (float*)d_out);
}
